// Multi_Task_Model_26766236189308
// MI455X (gfx1250) — matrix operand order repaired and checked by reading, compiled, not run on hardware
//
#include <hip/hip_runtime.h>
#include <hip/hip_bf16.h>

// ---------------------------------------------------------------------------
// Fused 4-layer MLP (384 -> 512 -> 216 -> 10 -> 2316), B = 16384, fp32 in/out.
// Strategy: output-bandwidth-bound (152 MB out @ 23.3 TB/s ~ 7.7 us floor), so
// fuse all layers per 16-row wave tile; GEMMs use v_wmma_f32_16x16x32_f16.
// Weights are pre-swizzled into WMMA B-fragment layout (f16) in d_ws.
// ---------------------------------------------------------------------------

typedef _Float16 v16h   __attribute__((ext_vector_type(16)));
typedef _Float16 half8  __attribute__((ext_vector_type(8)));
typedef float    v8f    __attribute__((ext_vector_type(8)));
typedef float    f32x4  __attribute__((ext_vector_type(4)));

#define DIM_IN   384
#define DIM_H1   512
#define DIM_H2   216
#define DIM_H2P  224     // padded to 14 tiles of 16
#define DIM_BOT  10
#define DIM_OUT  2316
#define KT1      12      // 384 / 32
#define NT1      32      // 512 / 16
#define KT2      16      // 512 / 32
#define NT2      14      // 224 / 16
#define KT3      7       // 224 / 32
#define NTH      145     // 2320 / 16

// Packed fragment sizes (in f16 elements): each frag = 32 lanes * 16 halfs = 512
#define W1P_ELEMS (NT1 * KT1 * 512)   // 196608
#define W2P_ELEMS (NT2 * KT2 * 512)   // 114688
#define W3P_ELEMS (1   * KT3 * 512)   //   3584
#define WHP_ELEMS (NTH * 1   * 512)   //  74240

// ---------------------------------------------------------------------------
// Pack fp32 weight W[Kr][Nr] (row-major, fan_in x fan_out) into f16 WMMA
// B-fragment layout. Fragment (nt,kt) -> index (nt*KT + kt). Within a frag:
// element (lane, i): N = nt*16 + (lane&15), K = kt*32 + (i>=8 ? 16 : 0) + (lane>=16 ? 8 : 0) + (i&7)
// (16-bit B 32x16: lanes 0-15 hold K {0..7, 16..23}, lanes 16-31 hold K {8..15, 24..31}.)
// Out-of-range K/N are zero-filled (pads contribute 0 to the accumulation).
// ---------------------------------------------------------------------------
__global__ void pack_w_kernel(const float* __restrict__ W, _Float16* __restrict__ dst,
                              int Kr, int Nr, int KT, int NT)
{
    int idx = blockIdx.x * blockDim.x + threadIdx.x;
    int total = NT * KT * 512;
    if (idx >= total) return;
    int e    = idx & 511;
    int frag = idx >> 9;
    int kt   = frag % KT;
    int nt   = frag / KT;
    int lane = e >> 4;
    int i    = e & 15;
    int n = nt * 16 + (lane & 15);
    int k = kt * 32 + ((i >> 3) << 4) + ((lane >> 4) << 3) + (i & 7);
    float v = (k < Kr && n < Nr) ? W[(size_t)k * Nr + n] : 0.0f;
    dst[idx] = (_Float16)v;
}

// ---------------------------------------------------------------------------
// Fused MLP. 64 threads = 2 waves per block; each wave owns 16 batch rows and
// runs all 4 layers. Intermediates live in per-wave LDS slices (no barriers:
// DS ops are in-order within a wave). EXEC stays full around every WMMA.
// ---------------------------------------------------------------------------
__global__ __launch_bounds__(64) void mlp_fused_kernel(
    const float*    __restrict__ x,
    const _Float16* __restrict__ W1p, const float* __restrict__ b1,
    const _Float16* __restrict__ W2p, const float* __restrict__ b2,
    const _Float16* __restrict__ W3p, const float* __restrict__ b3,
    const _Float16* __restrict__ Whp, const float* __restrict__ bh,
    float* __restrict__ out)
{
    __shared__ _Float16 sH1[2][16 * DIM_H1];    // 2 x 16 KB
    __shared__ _Float16 sH2[2][16 * DIM_H2P];   // 2 x  7 KB
    __shared__ _Float16 sF [2][16 * 16];        // 2 x 0.5 KB

    const int lane = threadIdx.x & 31;
    const int wv   = threadIdx.x >> 5;
    const int lrow = lane & 15;
    const int hi   = lane >> 4;                 // 0: lanes 0-15, 1: lanes 16-31
    const int row0 = (blockIdx.x * 2 + wv) * 16;

    // ---------------- Layer 1: [16x384] @ [384x512], bias + ReLU -> LDS ----
    // A fragments built straight from fp32 x with inline cvt to f16.
    // ISA 16-bit A 16x32: lane<16 holds K {0..7, 16..23}; lane>=16 {8..15, 24..31}.
    v16h A1[KT1];
    {
        const float* xr = x + (size_t)(row0 + lrow) * DIM_IN;
        #pragma unroll
        for (int kt = 0; kt < KT1; ++kt) {
            const int off = kt * 32 + hi * 8;
            f32x4 f0 = *(const f32x4*)(xr + off);
            f32x4 f1 = *(const f32x4*)(xr + off + 4);
            f32x4 f2 = *(const f32x4*)(xr + off + 16);
            f32x4 f3 = *(const f32x4*)(xr + off + 20);
            v16h a;
            #pragma unroll
            for (int i = 0; i < 4; ++i) {
                a[i]      = (_Float16)f0[i];
                a[4 + i]  = (_Float16)f1[i];
                a[8 + i]  = (_Float16)f2[i];
                a[12 + i] = (_Float16)f3[i];
            }
            A1[kt] = a;
        }
    }
    for (int nt = 0; nt < NT1; ++nt) {
        v8f acc = {};
        const _Float16* bp = W1p + (size_t)(nt * KT1) * 512 + lane * 16;
        #pragma unroll
        for (int kt = 0; kt < KT1; ++kt) {
            v16h bfrag = *(const v16h*)(bp + kt * 512);
            acc = __builtin_amdgcn_wmma_f32_16x16x32_f16(
                false, A1[kt], false, bfrag, (short)0, acc, false, false);
        }
        const int   col  = nt * 16 + lrow;
        const float bias = b1[col];
        #pragma unroll
        for (int r = 0; r < 8; ++r) {            // C layout: VGPR r -> M = r + 8*hi
            float v = acc[r] + bias;
            v = v > 0.0f ? v : 0.0f;
            sH1[wv][(r + hi * 8) * DIM_H1 + col] = (_Float16)v;
        }
    }

    // ---------------- Layer 2: [16x512] @ [512x216(pad224)], bias + ReLU ---
    for (int nt = 0; nt < NT2; ++nt) {
        v8f acc = {};
        const _Float16* bp = W2p + (size_t)(nt * KT2) * 512 + lane * 16;
        #pragma unroll
        for (int kt = 0; kt < KT2; ++kt) {
            const int off = kt * 32 + hi * 8;
            half8 lo = *(const half8*)&sH1[wv][lrow * DIM_H1 + off];
            half8 ho = *(const half8*)&sH1[wv][lrow * DIM_H1 + off + 16];
            v16h a;
            #pragma unroll
            for (int i = 0; i < 8; ++i) { a[i] = lo[i]; a[8 + i] = ho[i]; }
            v16h bfrag = *(const v16h*)(bp + kt * 512);
            acc = __builtin_amdgcn_wmma_f32_16x16x32_f16(
                false, a, false, bfrag, (short)0, acc, false, false);
        }
        const int   col  = nt * 16 + lrow;
        const float bias = (col < DIM_H2) ? b2[col] : 0.0f;   // pad cols -> 0
        #pragma unroll
        for (int r = 0; r < 8; ++r) {
            float v = acc[r] + bias;
            v = v > 0.0f ? v : 0.0f;
            sH2[wv][(r + hi * 8) * DIM_H2P + col] = (_Float16)v;
        }
    }

    // ---------------- Layer 3: [16x224] @ [224x16(10)], bias + ReLU --------
    {
        v8f acc = {};
        const _Float16* bp = W3p + lane * 16;
        #pragma unroll
        for (int kt = 0; kt < KT3; ++kt) {
            const int off = kt * 32 + hi * 8;
            half8 lo = *(const half8*)&sH2[wv][lrow * DIM_H2P + off];
            half8 ho = *(const half8*)&sH2[wv][lrow * DIM_H2P + off + 16];
            v16h a;
            #pragma unroll
            for (int i = 0; i < 8; ++i) { a[i] = lo[i]; a[8 + i] = ho[i]; }
            v16h bfrag = *(const v16h*)(bp + (size_t)kt * 512);
            acc = __builtin_amdgcn_wmma_f32_16x16x32_f16(
                false, a, false, bfrag, (short)0, acc, false, false);
        }
        const int   col  = lrow;
        const float bias = (col < DIM_BOT) ? b3[col] : 0.0f;  // pad cols -> 0
        #pragma unroll
        for (int r = 0; r < 8; ++r) {
            float v = acc[r] + bias;
            v = v > 0.0f ? v : 0.0f;
            sF[wv][(r + hi * 8) * 16 + col] = (_Float16)v;
        }
    }

    // ---------------- Head: [16x10(pad K=32)] @ [10x2316(pad 2320)] + bias -
    v16h A4;
    {
        // lane<16 needs K=0..7 (row cols 0..7); lane>=16 needs K=8..15.
        half8 lo = *(const half8*)&sF[wv][lrow * 16 + hi * 8];
        #pragma unroll
        for (int i = 0; i < 8; ++i) { A4[i] = lo[i]; A4[8 + i] = (_Float16)0.0f; }
    }
    for (int nt = 0; nt < NTH; ++nt) {
        const _Float16* bp = Whp + (size_t)nt * 512 + lane * 16;
        if (nt + 1 < NTH)
            __builtin_prefetch(bp + 512, 0, 0);   // global_prefetch_b8 next frag
        v16h bfrag = *(const v16h*)bp;
        v8f acc = {};
        acc = __builtin_amdgcn_wmma_f32_16x16x32_f16(
            false, A4, false, bfrag, (short)0, acc, false, false);
        const int  col   = nt * 16 + lrow;
        const bool valid = (col < DIM_OUT);
        const float bias = valid ? bh[col] : 0.0f;
        if (valid) {
            #pragma unroll
            for (int r = 0; r < 8; ++r)
                out[(size_t)(row0 + r + hi * 8) * DIM_OUT + col] = acc[r] + bias;
        }
    }
}

// ---------------------------------------------------------------------------
extern "C" void kernel_launch(void* const* d_in, const int* in_sizes, int n_in,
                              void* d_out, int out_size, void* d_ws, size_t ws_size,
                              hipStream_t stream) {
    const float* x  = (const float*)d_in[0];
    const float* W1 = (const float*)d_in[1];
    const float* b1 = (const float*)d_in[2];
    const float* W2 = (const float*)d_in[3];
    const float* b2 = (const float*)d_in[4];
    const float* W3 = (const float*)d_in[5];
    const float* b3 = (const float*)d_in[6];
    const float* Wh = (const float*)d_in[7];
    const float* bh = (const float*)d_in[8];
    float* out = (float*)d_out;

    _Float16* ws  = (_Float16*)d_ws;          // ~778 KB of f16 packed weights
    _Float16* W1p = ws;
    _Float16* W2p = W1p + W1P_ELEMS;
    _Float16* W3p = W2p + W2P_ELEMS;
    _Float16* Whp = W3p + W3P_ELEMS;

    // Pre-swizzle weights into WMMA B-fragment layout (deterministic each call).
    pack_w_kernel<<<(W1P_ELEMS + 255) / 256, 256, 0, stream>>>(W1, W1p, DIM_IN,  DIM_H1,  KT1, NT1);
    pack_w_kernel<<<(W2P_ELEMS + 255) / 256, 256, 0, stream>>>(W2, W2p, DIM_H1,  DIM_H2,  KT2, NT2);
    pack_w_kernel<<<(W3P_ELEMS + 255) / 256, 256, 0, stream>>>(W3, W3p, DIM_H2,  DIM_BOT, KT3, 1);
    pack_w_kernel<<<(WHP_ELEMS + 255) / 256, 256, 0, stream>>>(Wh, Whp, DIM_BOT, DIM_OUT, 1,   NTH);

    // 16384 rows / (2 waves * 16 rows) = 512 blocks of 64 threads (2 wave32s).
    mlp_fused_kernel<<<512, 64, 0, stream>>>(x, W1p, b1, W2p, b2, W3p, b3, Whp, bh, out);
}
